// BatchedDiffPool_88837103551465
// MI455X (gfx1250) — matrix operand order repaired and checked by reading, not compiled, not run on hardware
//
#include <hip/hip_runtime.h>

typedef __attribute__((ext_vector_type(16))) _Float16 v16h;
typedef __attribute__((ext_vector_type(8)))  _Float16 h8;
typedef __attribute__((ext_vector_type(8)))  float    v8f;

#define NN    8192   // nodes
#define FIN   256    // input features
#define NH    128    // hidden (embed)
#define NC    1024   // next-level clusters (assign)
#define LRA   0.2f   // leaky relu slope

// ---------------------------------------------------------------------------
// WMMA helpers (CDNA5 gfx1250, wave32, 16x16x32 f16 -> f32)
// ---------------------------------------------------------------------------
__device__ __forceinline__ v8f wmma_f16(v16h a, v16h b, v8f c) {
  // 8 args: (neg_a, A, neg_b, B, c_mod, C, reuse_a, reuse_b)
  return __builtin_amdgcn_wmma_f32_16x16x32_f16(false, a, false, b, (short)0, c,
                                                false, false);
}

// A-fragment (16x32 f16, row-major source, contiguous k order to match B):
// lane L: row = L&15, kb = (L<16)?0:16 ; elements 0..15 = K kb..kb+15
// -> two contiguous 16B loads.
__device__ __forceinline__ v16h load_a_frag(const _Float16* rowptr, int kbA) {
  h8 lo = *(const h8*)(rowptr + 2 * kbA);
  h8 hi = *(const h8*)(rowptr + 2 * kbA + 8);
  v16h a;
#pragma unroll
  for (int i = 0; i < 8; ++i) { a[i] = lo[i]; a[i + 8] = hi[i]; }
  return a;
}
// B-fragment (32x16 f16): lane L: col = L&15, K run = (L<16)?0..15:16..31,
// contiguous in a [Ncols, K] transposed operand -> single 32B load.

// ---------------------------------------------------------------------------
// Generic WMMA GEMM:  D = A(row-major [M,K]) x B  where B is given transposed
// as BT [Ncols, K] row-major.  TPW 16x16 tiles per wave along N.
// MODE 0: store f16 transposed  OUT[Ncols, M]
// MODE 1: store f32 row-major   OUT[M, Ncols]
// ---------------------------------------------------------------------------
template <int TPW, int MODE>
__global__ void __launch_bounds__(256)
gemm16(const _Float16* __restrict__ A, const _Float16* __restrict__ BT,
       void* __restrict__ out, int M, int Ncols, int K) {
  const int lane = threadIdx.x & 31;
  const int wave = blockIdx.x * (blockDim.x >> 5) + (threadIdx.x >> 5);
  const int ntb  = Ncols / (16 * TPW);
  const int mt   = wave / ntb;
  const int nt0  = (wave % ntb) * TPW;
  if (mt * 16 >= M) return;

  const int r = lane & 15;
  const int hi = lane >> 4;
  const int kbA = hi * 8, kbB = hi * 16, rbase = hi * 8;

  v8f acc[TPW] = {};
  const _Float16* arow = A + (size_t)(mt * 16 + r) * K;
  for (int k0 = 0; k0 < K; k0 += 32) {
    v16h a = load_a_frag(arow + k0, kbA);
#pragma unroll
    for (int t = 0; t < TPW; ++t) {
      v16h b = *(const v16h*)(BT + (size_t)((nt0 + t) * 16 + r) * K + k0 + kbB);
      acc[t] = wmma_f16(a, b, acc[t]);
    }
  }

  if constexpr (MODE == 0) {
    _Float16* O = (_Float16*)out;  // [Ncols, M]
#pragma unroll
    for (int t = 0; t < TPW; ++t) {
      int colg = (nt0 + t) * 16 + r;
      h8 o;
#pragma unroll
      for (int v = 0; v < 8; ++v) o[v] = (_Float16)acc[t][v];
      *(h8*)(O + (size_t)colg * M + mt * 16 + rbase) = o;
    }
  } else {
    float* O = (float*)out;  // [M, Ncols]
#pragma unroll
    for (int t = 0; t < TPW; ++t) {
      int colg = (nt0 + t) * 16 + r;
#pragma unroll
      for (int v = 0; v < 8; ++v)
        O[(size_t)(mt * 16 + rbase + v) * Ncols + colg] = acc[t][v];
    }
  }
}

// ---------------------------------------------------------------------------
// Masked softmax-weighted aggregation (flash-style, no NxN materialization):
//   out[i,:] = sum_j w_ij * B[j,:] ,
//   ATT:  w_ij = (adj_ij>0) ? exp(lrelu(f1_i+f2_j) - m_i) : 0   (scaled 1/Z_i)
//   BIN:  w_ij = (adj_ij>0) ? 1 : 0
// BT = value matrix transposed [FOUT, NN] f16.
// EPI 0: elu(acc/Z + h) -> f16 transposed OUT[FOUT, NN]   (z_l^T)
// EPI 1: elu(acc/Z + h) -> f32 row-major  OUT[NN, FOUT]   (assign pre-softmax)
// EPI 2: acc            -> f16 transposed OUT[FOUT, NN]   (t^T = (adj@s_l)^T)
// One block per 16 rows; 8 waves split FOUT.
// ---------------------------------------------------------------------------
template <int FOUT, bool BIN, int EPI>
__global__ void __launch_bounds__(256)
pv16(const float* __restrict__ adj, const float* __restrict__ f1,
     const float* __restrict__ f2, const float* __restrict__ mrow,
     const float* __restrict__ zrow, const _Float16* __restrict__ BT,
     const _Float16* __restrict__ Hadd, void* __restrict__ out) {
  constexpr int TPW = FOUT / 8 / 16;     // tiles per wave (1 or 8)
  __shared__ __align__(16) _Float16 wt[16 * 40];  // 16x32 tile, padded rows
  __shared__ float sf1[16], sm[16];

  const int tid = threadIdx.x;
  const int i0 = blockIdx.x * 16;
  if constexpr (!BIN) {
    if (tid < 16) { sf1[tid] = f1[i0 + tid]; sm[tid] = mrow[i0 + tid]; }
  }
  const int lane = tid & 31;
  const int wv = tid >> 5;  // 0..7
  const int r = lane & 15;
  const int hi = lane >> 4;
  const int kbA = hi * 8, kbB = hi * 16, rbase = hi * 8;
  const int c0 = wv * (FOUT / 8);

  v8f acc[TPW] = {};
  __syncthreads();

  for (int k0 = 0; k0 < NN; k0 += 32) {
    // cooperatively build the 16x32 f16 weight tile
#pragma unroll
    for (int e = 0; e < 2; ++e) {
      int idx = tid + e * 256;
      int rr = idx >> 5, kk = idx & 31;
      int j = k0 + kk;
      const float* ap = adj + (size_t)(i0 + rr) * NN + j;
      float av = *ap;
      if (k0 + 256 < NN) __builtin_prefetch(ap + 256, 0, 1);
      float wvv;
      if constexpr (BIN) {
        wvv = av > 0.f ? 1.f : 0.f;
      } else {
        wvv = 0.f;
        if (av > 0.f) {
          float ee = sf1[rr] + f2[j];
          ee = ee > 0.f ? ee : LRA * ee;
          wvv = __expf(ee - sm[rr]);
        }
      }
      wt[rr * 40 + kk] = (_Float16)wvv;
    }
    __syncthreads();
    v16h a = load_a_frag(&wt[r * 40], kbA);
#pragma unroll
    for (int t = 0; t < TPW; ++t) {
      v16h b = *(const v16h*)(BT + (size_t)(c0 + t * 16 + r) * NN + k0 + kbB);
      acc[t] = wmma_f16(a, b, acc[t]);
    }
    __syncthreads();
  }

  if constexpr (EPI == 0) {
    _Float16* O = (_Float16*)out;  // z_l^T [FOUT, NN]
#pragma unroll
    for (int t = 0; t < TPW; ++t) {
      int colg = c0 + t * 16 + r;
      h8 hv = *(const h8*)(Hadd + (size_t)colg * NN + i0 + rbase);
      h8 o;
#pragma unroll
      for (int v = 0; v < 8; ++v) {
        float zz = zrow[i0 + rbase + v];
        zz = zz > 0.f ? zz : 1.f;
        float val = acc[t][v] / zz + (float)hv[v];
        val = val > 0.f ? val : __expf(val) - 1.f;  // elu
        o[v] = (_Float16)val;
      }
      *(h8*)(O + (size_t)colg * NN + i0 + rbase) = o;
    }
  } else if constexpr (EPI == 1) {
    float* O = (float*)out;  // assign [NN, FOUT]
#pragma unroll
    for (int t = 0; t < TPW; ++t) {
      int colg = c0 + t * 16 + r;
      h8 hv = *(const h8*)(Hadd + (size_t)colg * NN + i0 + rbase);
#pragma unroll
      for (int v = 0; v < 8; ++v) {
        float zz = zrow[i0 + rbase + v];
        zz = zz > 0.f ? zz : 1.f;
        float val = acc[t][v] / zz + (float)hv[v];
        val = val > 0.f ? val : __expf(val) - 1.f;
        O[(size_t)(i0 + rbase + v) * FOUT + colg] = val;
      }
    }
  } else {
    _Float16* O = (_Float16*)out;  // t^T [FOUT, NN]
#pragma unroll
    for (int t = 0; t < TPW; ++t) {
      int colg = c0 + t * 16 + r;
      h8 o;
#pragma unroll
      for (int v = 0; v < 8; ++v) o[v] = (_Float16)acc[t][v];
      *(h8*)(O + (size_t)colg * NN + i0 + rbase) = o;
    }
  }
}

// ---------------------------------------------------------------------------
// Per-row online-softmax stats for BOTH GAT heads in one pass over adj.
// m_i = max over adjacent j of lrelu(f1_i+f2_j);  Z_i = sum exp(e - m).
// ---------------------------------------------------------------------------
__global__ void __launch_bounds__(256)
stats_kernel(const float* __restrict__ adj, const float* __restrict__ f1e,
             const float* __restrict__ f2e, const float* __restrict__ f1a,
             const float* __restrict__ f2a, float* __restrict__ me,
             float* __restrict__ ze, float* __restrict__ ma,
             float* __restrict__ za) {
  const int i = blockIdx.x;
  const int tid = threadIdx.x;
  const float fe = f1e[i], fa = f1a[i];
  float m1 = -1e30f, z1 = 0.f, m2 = -1e30f, z2 = 0.f;
  const float* arow = adj + (size_t)i * NN;
  for (int j = tid; j < NN; j += 256) {
    float av = arow[j];
    if (av > 0.f) {
      float e1 = fe + f2e[j];
      e1 = e1 > 0.f ? e1 : LRA * e1;
      if (e1 > m1) { z1 = z1 * __expf(m1 - e1) + 1.f; m1 = e1; }
      else z1 += __expf(e1 - m1);
      float e2 = fa + f2a[j];
      e2 = e2 > 0.f ? e2 : LRA * e2;
      if (e2 > m2) { z2 = z2 * __expf(m2 - e2) + 1.f; m2 = e2; }
      else z2 += __expf(e2 - m2);
    }
  }
  __shared__ float s_m1[256], s_z1[256], s_m2[256], s_z2[256];
  s_m1[tid] = m1; s_z1[tid] = z1; s_m2[tid] = m2; s_z2[tid] = z2;
  __syncthreads();
  for (int s = 128; s > 0; s >>= 1) {
    if (tid < s) {
      float M1 = fmaxf(s_m1[tid], s_m1[tid + s]);
      s_z1[tid] = s_z1[tid] * __expf(s_m1[tid] - M1) +
                  s_z1[tid + s] * __expf(s_m1[tid + s] - M1);
      s_m1[tid] = M1;
      float M2 = fmaxf(s_m2[tid], s_m2[tid + s]);
      s_z2[tid] = s_z2[tid] * __expf(s_m2[tid] - M2) +
                  s_z2[tid + s] * __expf(s_m2[tid + s] - M2);
      s_m2[tid] = M2;
    }
    __syncthreads();
  }
  if (tid == 0) { me[i] = s_m1[0]; ze[i] = s_z1[0]; ma[i] = s_m2[0]; za[i] = s_z2[0]; }
}

// f1_i = h[i,:].a[:F], f2_i = h[i,:].a[F:]  from transposed f16 h.
template <int FOUT>
__global__ void __launch_bounds__(256)
f12_kernel(const _Float16* __restrict__ hT, const float* __restrict__ avec,
           float* __restrict__ f1, float* __restrict__ f2) {
  int i = blockIdx.x * 256 + threadIdx.x;
  float a1 = 0.f, a2 = 0.f;
  for (int c = 0; c < FOUT; ++c) {
    float v = (float)hT[(size_t)c * NN + i];
    a1 += v * avec[c];
    a2 += v * avec[FOUT + c];
  }
  f1[i] = a1; f2[i] = a2;
}

// Row softmax of assign -> s_l f32 (d_out slice) + s_l^T f16 for GEMMs.
__global__ void __launch_bounds__(256)
softmax_kernel(const float* __restrict__ assign, float* __restrict__ sl,
               _Float16* __restrict__ slT) {
  const int i = blockIdx.x;
  const int tid = threadIdx.x;
  float vals[NC / 256];
  float mx = -1e30f;
#pragma unroll
  for (int e = 0; e < NC / 256; ++e) {
    vals[e] = assign[(size_t)i * NC + tid + e * 256];
    mx = fmaxf(mx, vals[e]);
  }
  __shared__ float red[256];
  red[tid] = mx;
  __syncthreads();
  for (int s = 128; s > 0; s >>= 1) {
    if (tid < s) red[tid] = fmaxf(red[tid], red[tid + s]);
    __syncthreads();
  }
  mx = red[0];
  __syncthreads();
  float sum = 0.f;
#pragma unroll
  for (int e = 0; e < NC / 256; ++e) { vals[e] = __expf(vals[e] - mx); sum += vals[e]; }
  red[tid] = sum;
  __syncthreads();
  for (int s = 128; s > 0; s >>= 1) {
    if (tid < s) red[tid] += red[tid + s];
    __syncthreads();
  }
  float inv = 1.f / red[0];
#pragma unroll
  for (int e = 0; e < NC / 256; ++e) {
    int c = tid + e * 256;
    float sv = vals[e] * inv;
    sl[(size_t)i * NC + c] = sv;
    slT[(size_t)c * NN + i] = (_Float16)sv;
  }
}

__global__ void __launch_bounds__(256)
cvt_f32_f16(const float* __restrict__ in, _Float16* __restrict__ out, int n) {
  int i = blockIdx.x * 256 + threadIdx.x;
  if (i < n) out[i] = (_Float16)in[i];
}

__global__ void __launch_bounds__(256)
cvt_transpose_f16(const float* __restrict__ W, _Float16* __restrict__ WT,
                  int rows, int cols) {
  int idx = blockIdx.x * 256 + threadIdx.x;
  if (idx < rows * cols) {
    int rr = idx / cols, cc = idx % cols;
    WT[(size_t)cc * rows + rr] = (_Float16)W[idx];
  }
}

// ---------------------------------------------------------------------------
extern "C" void kernel_launch(void* const* d_in, const int* in_sizes, int n_in,
                              void* d_out, int out_size, void* d_ws,
                              size_t ws_size, hipStream_t stream) {
  (void)in_sizes; (void)n_in; (void)out_size; (void)ws_size;
  const float* x   = (const float*)d_in[0];
  const float* adj = (const float*)d_in[1];
  const float* We  = (const float*)d_in[2];
  const float* ae  = (const float*)d_in[3];
  const float* Wa  = (const float*)d_in[4];
  const float* aa  = (const float*)d_in[5];

  float* out   = (float*)d_out;
  float* xnext = out;                                   // [NC, NH]
  float* anext = out + (size_t)NC * NH;                 // [NC, NC]
  float* sl    = out + (size_t)NC * NH + (size_t)NC * NC;  // [NN, NC]

  char* wp = (char*)d_ws;
  auto alloc = [&](size_t bytes) -> char* {
    char* p = wp;
    wp += (bytes + 255) & ~(size_t)255;
    return p;
  };
  _Float16* x16 = (_Float16*)alloc((size_t)NN * FIN * 2);
  _Float16* WeT = (_Float16*)alloc((size_t)NH * FIN * 2);
  _Float16* WaT = (_Float16*)alloc((size_t)NC * FIN * 2);
  _Float16* heT = (_Float16*)alloc((size_t)NH * NN * 2);   // h_embed^T
  _Float16* haT = (_Float16*)alloc((size_t)NC * NN * 2);   // h_assign^T
  float* f1e = (float*)alloc(NN * 4);
  float* f2e = (float*)alloc(NN * 4);
  float* f1a = (float*)alloc(NN * 4);
  float* f2a = (float*)alloc(NN * 4);
  float* me  = (float*)alloc(NN * 4);
  float* ze  = (float*)alloc(NN * 4);
  float* ma  = (float*)alloc(NN * 4);
  float* za  = (float*)alloc(NN * 4);
  _Float16* zlT    = (_Float16*)alloc((size_t)NH * NN * 2);  // z_l^T
  float*    assign = (float*)alloc((size_t)NN * NC * 4);
  _Float16* slT    = (_Float16*)alloc((size_t)NC * NN * 2);  // s_l^T
  _Float16* tT     = (_Float16*)alloc((size_t)NC * NN * 2);  // (adj@s_l)^T

  // 1) fp32 -> f16 conversions (W transposed for B-operand layout)
  cvt_f32_f16<<<(NN * FIN + 255) / 256, 256, 0, stream>>>(x, x16, NN * FIN);
  cvt_transpose_f16<<<(FIN * NH + 255) / 256, 256, 0, stream>>>(We, WeT, FIN, NH);
  cvt_transpose_f16<<<(FIN * NC + 255) / 256, 256, 0, stream>>>(Wa, WaT, FIN, NC);

  // 2) h = x @ W  (stored transposed f16)
  {
    int waves = (NN / 16) * (NH / 64);
    gemm16<4, 0><<<(waves + 7) / 8, 256, 0, stream>>>(x16, WeT, heT, NN, NH, FIN);
    waves = (NN / 16) * (NC / 64);
    gemm16<4, 0><<<(waves + 7) / 8, 256, 0, stream>>>(x16, WaT, haT, NN, NC, FIN);
  }

  // 3) attention projections f1/f2 for both heads
  f12_kernel<NH><<<NN / 256, 256, 0, stream>>>(heT, ae, f1e, f2e);
  f12_kernel<NC><<<NN / 256, 256, 0, stream>>>(haT, aa, f1a, f2a);

  // 4) per-row softmax stats for both heads (single pass over adj)
  stats_kernel<<<NN, 256, 0, stream>>>(adj, f1e, f2e, f1a, f2a, me, ze, ma, za);

  // 5) z_l = elu(att_e @ h_e + h_e)   -> z_l^T f16
  pv16<NH, false, 0><<<NN / 16, 256, 0, stream>>>(adj, f1e, f2e, me, ze, heT, heT, zlT);

  // 6) assign = elu(att_a @ h_a + h_a) -> f32 row-major
  pv16<NC, false, 1><<<NN / 16, 256, 0, stream>>>(adj, f1a, f2a, ma, za, haT, haT, assign);

  // 7) s_l = softmax(assign)  -> d_out slice (f32) + s_l^T f16
  softmax_kernel<<<NN, 256, 0, stream>>>(assign, sl, slT);

  // 8) t^T = (adj @ s_l)^T  (binary-weight WMMA pass over adj)
  pv16<NC, true, 2><<<NN / 16, 256, 0, stream>>>(adj, nullptr, nullptr, nullptr,
                                                 nullptr, slT, nullptr, tT);

  // 9) xnext = s_l^T @ z_l ;  anext = s_l^T @ t  (f32 outputs)
  {
    int waves = (NC / 16) * (NH / 64);
    gemm16<4, 1><<<(waves + 7) / 8, 256, 0, stream>>>(slT, zlT, xnext, NC, NH, NN);
    waves = (NC / 16) * (NC / 64);
    gemm16<4, 1><<<(waves + 7) / 8, 256, 0, stream>>>(slT, tT, anext, NC, NC, NN);
  }
}
